// Transformer3DBlock_13529146982800
// MI455X (gfx1250) — matrix operand order repaired and checked by reading, not compiled, not run on hardware
//
#include <hip/hip_runtime.h>
#include <hip/hip_bf16.h>
#include <math.h>

// ---------------------------------------------------------------------------
// CDNA5 / gfx1250 Swin-3D transformer block.
//  - all matmuls: v_wmma_f32_16x16x32_f16 (f16 in, f32 accumulate)
//  - GEMM tiles staged with GLOBAL_LOAD_ASYNC_TO_LDS_B128 (ASYNCcnt) and
//    double-buffered LDS: next K-slice loads overlap current WMMAs.
//  - wave32: block of 128 threads == 4 waves.
// ---------------------------------------------------------------------------

typedef _Float16 half_t;
typedef __attribute__((ext_vector_type(16))) _Float16 v16h;
typedef __attribute__((ext_vector_type(8)))  float    v8f;
typedef __attribute__((ext_vector_type(4)))  unsigned int u32x4;

union FragH { v16h h; u32x4 u[2]; };

// async copy global->LDS, 16B per lane, bypasses VGPRs (ISA 15.18.3 op 98)
#define ASYNC_LD_B128(ldsOff, gPtr)                                         \
  asm volatile("global_load_async_to_lds_b128 %0, %1, off"                  \
               :: "v"(ldsOff), "v"((unsigned long long)(uintptr_t)(gPtr))   \
               : "memory")
#define WAIT_ASYNC0() asm volatile("s_wait_asynccnt 0x0" ::: "memory")

// Problem constants
#define LTOK   138240      // 8*96*180 tokens
#define CDIM   192
#define NHEAD  6
#define DHEAD  32
#define NWIN   144         // tokens per window (2*6*12)
#define NWINP  160         // padded K for P@V
#define NW     64          // windows per lon-slice (4*16)
#define NLON   15
#define HDIM   768
#define MCHUNK 27648       // L/5 token chunk for MLP

// ---------------------------------------------------------------------------
// EPI / shift-window mask are pure functions of token coords -> recompute.
// ---------------------------------------------------------------------------
__device__ __forceinline__ int epi_index(int n, int m) {
  int zn = n / 72, rn = n % 72, hn = rn / 12, wn = rn % 12;
  int zm = m / 72, rm = m % 72, hm = rm / 12, wm = rm % 12;
  return (zn + zm * 2) * 828 + (hn + hm * 6) * 23 + (wn - wm + 11);
}

__device__ __forceinline__ int region_code(int lw, int w, int tok) {
  int zp = tok / 72, r = tok % 72, zl = r / 12, zo = r % 12;
  int pr = (w >> 4) * 2 + zp;
  int lr = (w & 15) * 6 + zl;
  int oc = lw * 12 + zo;
  int pi = (pr < 6) ? 0 : (pr < 7 ? 1 : 2);
  int li = (lr < 90) ? 0 : (lr < 93 ? 1 : 2);
  int oi = (oc < 174) ? 0 : 1;
  return pi * 9 + li * 3 + oi;
}

// windowed token index -> original flat L index (roll by +shift)
__device__ __forceinline__ size_t win_to_orig(int t) {
  int bw = t / NWIN, tok = t % NWIN;
  int lw = bw / NW, w = bw % NW;
  int zp = tok / 72, r = tok % 72, zl = r / 12, zo = r % 12;
  int pr = (w >> 4) * 2 + zp;
  int lr = (w & 15) * 6 + zl;
  int oc = lw * 12 + zo;
  int ps = pr + 1;  if (ps >= 8)   ps -= 8;
  int ls = lr + 3;  if (ls >= 96)  ls -= 96;
  int os = oc + 6;  if (os >= 180) os -= 180;
  return ((size_t)ps * 96 + ls) * 180 + os;
}

// ---------------------------------------------------------------------------
// weight convert + transpose: src f32 [K][N] row-major -> dst f16 [N][K]
// ---------------------------------------------------------------------------
__global__ void convT_kernel(const float* __restrict__ src, half_t* __restrict__ dst,
                             int K, int N) {
  int i = blockIdx.x * 256 + threadIdx.x;
  if (i >= K * N) return;
  int n = i / K, k = i % K;
  dst[i] = (half_t)src[(size_t)k * N + n];
}

// ---------------------------------------------------------------------------
// LayerNorm, one wave per token, 6 channels/lane; shifted!=0 gathers through
// the roll + window-partition permutation (norm1 path).
// ---------------------------------------------------------------------------
__global__ __launch_bounds__(256)
void ln_kernel(const float* __restrict__ xin, const float* __restrict__ g,
               const float* __restrict__ b, half_t* __restrict__ out, int shifted) {
  int wave = threadIdx.x >> 5, lane = threadIdx.x & 31;
  int t = blockIdx.x * 8 + wave;
  size_t src = shifted ? win_to_orig(t) : (size_t)t;
  float xv[6];
  float s = 0.f, s2 = 0.f;
#pragma unroll
  for (int j = 0; j < 6; ++j) {
    xv[j] = xin[src * CDIM + lane + j * 32];
    s += xv[j]; s2 += xv[j] * xv[j];
  }
#pragma unroll
  for (int off = 16; off; off >>= 1) {
    s  += __shfl_xor(s,  off, 32);
    s2 += __shfl_xor(s2, off, 32);
  }
  float mean = s * (1.f / CDIM);
  float var  = s2 * (1.f / CDIM) - mean * mean;
  float rstd = rsqrtf(var + 1e-5f);
#pragma unroll
  for (int j = 0; j < 6; ++j) {
    int c = lane + j * 32;
    out[(size_t)t * CDIM + c] = (half_t)((xv[j] - mean) * rstd * g[c] + b[c]);
  }
}

// ---------------------------------------------------------------------------
// WMMA GEMM: A f16 [M][K] row-major, BT f16 [N][K] row-major.
// 128 threads / 4 waves, block tile 128x64, wave tile 32x64 (2x4 WMMA tiles),
// K-step 32, LDS double-buffered via async-to-LDS copies.
// MODE 0: QKV (scale q, scatter per (window,head))    MODE 1: proj + unroll + resid
// MODE 2: fc1 + exact GELU -> f16                      MODE 3: fc2 + resid -> f32
// ---------------------------------------------------------------------------
template <int MODE>
__global__ __launch_bounds__(128)
void gemm128_kernel(const half_t* __restrict__ A, const half_t* __restrict__ BT,
                    const float* __restrict__ bias, int K,
                    float* __restrict__ fout,
                    half_t* __restrict__ h0, half_t* __restrict__ h1,
                    half_t* __restrict__ h2,
                    const float* __restrict__ resid, int m_off) {
  __shared__ __align__(16) half_t As[2][128 * 32];   // 2 x 8 KB
  __shared__ __align__(16) half_t Bs[2][64 * 32];    // 2 x 4 KB
  const int tid  = threadIdx.x;
  const int wave = tid >> 5;
  const int lane = tid & 31;
  const int mblk = blockIdx.x * 128;
  const int nblk = blockIdx.y * 64;

  v8f acc[2][4];
#pragma unroll
  for (int mi = 0; mi < 2; ++mi)
#pragma unroll
    for (int t = 0; t < 4; ++t) acc[mi][t] = (v8f){};

  // issue one K-slice of async global->LDS copies (4 A + 2 B b128 per thread)
  auto issue = [&](int k0, int b) {
    unsigned aBase = (unsigned)(uintptr_t)(&As[b][0]);
    unsigned bBase = (unsigned)(uintptr_t)(&Bs[b][0]);
#pragma unroll
    for (int j = 0; j < 4; ++j) {
      int chunk = j * 128 + tid;              // 0..511 : 128 rows x 4 x 16B
      int row = chunk >> 2;
      int col = (chunk & 3) * 8;              // halves
      unsigned lds = aBase + (unsigned)(row * 64 + col * 2);
      ASYNC_LD_B128(lds, A + (size_t)(mblk + row) * K + k0 + col);
    }
#pragma unroll
    for (int j = 0; j < 2; ++j) {
      int chunk = j * 128 + tid;              // 0..255 : 64 rows x 4 x 16B
      int row = chunk >> 2;
      int col = (chunk & 3) * 8;
      unsigned lds = bBase + (unsigned)(row * 64 + col * 2);
      ASYNC_LD_B128(lds, BT + (size_t)(nblk + row) * K + k0 + col);
    }
  };

  issue(0, 0);
  WAIT_ASYNC0();
  __syncthreads();

  int buf = 0;
  for (int k0 = 0; k0 < K; k0 += 32) {
    if (k0 + 32 < K) issue(k0 + 32, buf ^ 1);   // overlap next loads with WMMA

    const half_t* as = &As[buf][0];
    const half_t* bs = &Bs[buf][0];
    FragH afr[2];
    const int ab = (lane >> 4) * 8;
#pragma unroll
    for (int mi = 0; mi < 2; ++mi) {
      const int am = wave * 32 + mi * 16 + (lane & 15);
      afr[mi].u[0] = *(const u32x4*)(as + am * 32 + ab);
      afr[mi].u[1] = *(const u32x4*)(as + am * 32 + 16 + ab);
    }
#pragma unroll
    for (int t = 0; t < 4; ++t) {
      FragH bfr;
      const int bn = t * 16 + (lane & 15);
      const int kb = (lane >> 4) * 8;
      bfr.u[0] = *(const u32x4*)(bs + bn * 32 + kb);
      bfr.u[1] = *(const u32x4*)(bs + bn * 32 + kb + 16);
#pragma unroll
      for (int mi = 0; mi < 2; ++mi)
        acc[mi][t] = __builtin_amdgcn_wmma_f32_16x16x32_f16(
            false, afr[mi].h, false, bfr.h, (short)0, acc[mi][t], false, false);
    }
    WAIT_ASYNC0();       // next buffer resident
    __syncthreads();     // all waves done reading current buffer
    buf ^= 1;
  }

  // ---- epilogue -----------------------------------------------------------
#pragma unroll
  for (int mi = 0; mi < 2; ++mi) {
#pragma unroll
    for (int t = 0; t < 4; ++t) {
      const int n = nblk + t * 16 + (lane & 15);
      const float bv = bias[n];
#pragma unroll
      for (int vv = 0; vv < 8; ++vv) {
        const int m = mblk + wave * 32 + mi * 16 + (lane >> 4) * 8 + vv;
        float val = acc[mi][t][vv] + bv;
        if (MODE == 0) {                       // QKV scatter
          int which = n / CDIM, rem = n % CDIM;
          int head = rem >> 5, dch = rem & 31;
          int bw = m / NWIN, tok = m % NWIN;
          size_t idx = (((size_t)(bw * NHEAD + head)) * NWIN + tok) * DHEAD + dch;
          if (which == 0)      h0[idx] = (half_t)(val * 0.1767766952966369f);
          else if (which == 1) h1[idx] = (half_t)val;
          else                 h2[idx] = (half_t)val;
        } else if (MODE == 1) {                // proj: un-window + roll + resid
          size_t Lidx = win_to_orig(m);
          fout[Lidx * CDIM + n] = val + resid[Lidx * CDIM + n];
        } else if (MODE == 2) {                // fc1 + exact GELU
          float gl = 0.5f * val * (1.0f + erff(val * 0.70710678118654752f));
          h0[(size_t)m * HDIM + n] = (half_t)gl;
        } else {                               // fc2 + residual -> f32 out
          size_t mg = (size_t)(m + m_off);
          fout[mg * CDIM + n] = val + resid[mg * CDIM + n];
        }
      }
    }
  }
}

// ---------------------------------------------------------------------------
// Attention: one block (4 waves) per (window-batch, head).
// QK^T: 81 tiles, K=32 -> single WMMA each. f32 softmax over f16 LDS scores.
// P@V: 18 tiles, K zero-padded to 160, 5 WMMA steps.
// ---------------------------------------------------------------------------
__global__ __launch_bounds__(128)
void attn_kernel(const half_t* __restrict__ q, const half_t* __restrict__ k,
                 const half_t* __restrict__ v,
                 const float* __restrict__ bias_table,
                 half_t* __restrict__ out) {
  __shared__ __align__(16) half_t Ss[NWIN * NWINP];    // 46080 B
  __shared__ __align__(16) half_t vTs[DHEAD * NWINP];  // 10240 B

  const int bwh  = blockIdx.x;
  const int bw   = bwh / NHEAD, head = bwh % NHEAD;
  const int lw   = bw / NW, w = bw % NW;
  const int tid  = threadIdx.x;
  const int wave = tid >> 5;
  const int lane = tid & 31;

  const half_t* qb = q + (size_t)bwh * NWIN * DHEAD;
  const half_t* kb = k + (size_t)bwh * NWIN * DHEAD;
  const half_t* vb = v + (size_t)bwh * NWIN * DHEAD;
  __builtin_prefetch(qb + tid * 64, 0, 3);   // global_prefetch_b8
  __builtin_prefetch(kb + tid * 64, 0, 3);

  // v transposed into LDS (B^T layout for P@V), zero-padded K
  for (int i = tid; i < DHEAD * NWINP; i += 128) {
    int d = i / NWINP, t = i % NWINP;
    vTs[i] = (t < NWIN) ? vb[t * DHEAD + d] : (half_t)0.f;
  }

  // ---- S = (q*scale) @ k^T + bias + mask ----------------------------------
  for (int tt = wave; tt < 81; tt += 4) {
    const int ti = tt / 9, tj = tt % 9;
    FragH afr, bfr;
    const int am = ti * 16 + (lane & 15);
    const int ab = (lane >> 4) * 8;
    afr.u[0] = *(const u32x4*)(qb + am * DHEAD + ab);
    afr.u[1] = *(const u32x4*)(qb + am * DHEAD + 16 + ab);
    const int bn  = tj * 16 + (lane & 15);
    const int kb2 = (lane >> 4) * 8;
    bfr.u[0] = *(const u32x4*)(kb + bn * DHEAD + kb2);
    bfr.u[1] = *(const u32x4*)(kb + bn * DHEAD + kb2 + 16);
    v8f sacc = (v8f){};
    sacc = __builtin_amdgcn_wmma_f32_16x16x32_f16(
        false, afr.h, false, bfr.h, (short)0, sacc, false, false);

    const int mcol = tj * 16 + (lane & 15);
    const int cm = region_code(lw, w, mcol);
#pragma unroll
    for (int vv = 0; vv < 8; ++vv) {
      const int nrow = ti * 16 + (lane >> 4) * 8 + vv;
      const int cn = region_code(lw, w, nrow);
      const int epi = epi_index(nrow, mcol);
      float val = sacc[vv]
                + bias_table[((size_t)epi * NW + w) * NHEAD + head]
                + ((cn == cm) ? 0.f : -100.f);
      Ss[nrow * NWINP + mcol] = (half_t)val;
    }
  }
  __syncthreads();

  // ---- softmax rows (f32 math, f16 storage), zero pad cols ----------------
  for (int r = tid; r < NWIN; r += 128) {
    half_t* row = Ss + r * NWINP;
    float mx = -1e30f;
    for (int c = 0; c < NWIN; ++c) mx = fmaxf(mx, (float)row[c]);
    float sum = 0.f;
    for (int c = 0; c < NWIN; ++c) {
      float e = __expf((float)row[c] - mx);
      sum += e;
      row[c] = (half_t)e;
    }
    float inv = 1.f / sum;
    for (int c = 0; c < NWIN; ++c) row[c] = (half_t)((float)row[c] * inv);
    for (int c = NWIN; c < NWINP; ++c) row[c] = (half_t)0.f;
  }
  __syncthreads();

  // ---- out = P @ v --------------------------------------------------------
  for (int tt = wave; tt < 18; tt += 4) {
    const int ti = tt >> 1, tj = tt & 1;
    v8f oacc = (v8f){};
    const int am = ti * 16 + (lane & 15);
    const int ab = (lane >> 4) * 8;
    const int bn  = tj * 16 + (lane & 15);
    const int kb2 = (lane >> 4) * 8;
#pragma unroll
    for (int k0 = 0; k0 < NWINP; k0 += 32) {
      FragH afr, bfr;
      afr.u[0] = *(const u32x4*)(Ss + am * NWINP + k0 + ab);
      afr.u[1] = *(const u32x4*)(Ss + am * NWINP + k0 + 16 + ab);
      bfr.u[0] = *(const u32x4*)(vTs + bn * NWINP + k0 + kb2);
      bfr.u[1] = *(const u32x4*)(vTs + bn * NWINP + k0 + kb2 + 16);
      oacc = __builtin_amdgcn_wmma_f32_16x16x32_f16(
          false, afr.h, false, bfr.h, (short)0, oacc, false, false);
    }
    const int dcol = tj * 16 + (lane & 15);
#pragma unroll
    for (int vv = 0; vv < 8; ++vv) {
      const int row = ti * 16 + (lane >> 4) * 8 + vv;
      out[((size_t)bw * NWIN + row) * CDIM + head * DHEAD + dcol] = (half_t)oacc[vv];
    }
  }
}

// ---------------------------------------------------------------------------
// host launch
// ---------------------------------------------------------------------------
extern "C" void kernel_launch(void* const* d_in, const int* in_sizes, int n_in,
                              void* d_out, int out_size, void* d_ws, size_t ws_size,
                              hipStream_t stream) {
  const float* x       = (const float*)d_in[0];
  const float* n1g     = (const float*)d_in[1];
  const float* n1b     = (const float*)d_in[2];
  const float* qkv_w   = (const float*)d_in[3];
  const float* qkv_b   = (const float*)d_in[4];
  const float* btab    = (const float*)d_in[5];
  const float* proj_w  = (const float*)d_in[6];
  const float* proj_b  = (const float*)d_in[7];
  const float* n2g     = (const float*)d_in[8];
  const float* n2b     = (const float*)d_in[9];
  const float* fc1_w   = (const float*)d_in[10];
  const float* fc1_b   = (const float*)d_in[11];
  const float* fc2_w   = (const float*)d_in[12];
  const float* fc2_b   = (const float*)d_in[13];
  float* out = (float*)d_out;
  char* ws = (char*)d_ws;

  size_t off = 0;
  auto carve = [&](size_t bytes) { size_t r = off; off += (bytes + 255) & ~(size_t)255; return r; };
  half_t* qkvT = (half_t*)(ws + carve((size_t)576 * 192 * 2));
  half_t* projT = (half_t*)(ws + carve((size_t)192 * 192 * 2));
  half_t* fc1T = (half_t*)(ws + carve((size_t)768 * 192 * 2));
  half_t* fc2T = (half_t*)(ws + carve((size_t)192 * 768 * 2));
  half_t* A1   = (half_t*)(ws + carve((size_t)LTOK * CDIM * 2));   // also attn_out
  half_t* qbuf = (half_t*)(ws + carve((size_t)LTOK * CDIM * 2));   // also A2
  half_t* kbuf = (half_t*)(ws + carve((size_t)LTOK * CDIM * 2));
  half_t* vbuf = (half_t*)(ws + carve((size_t)LTOK * CDIM * 2));
  float*  xmid = (float*)(ws + carve((size_t)LTOK * CDIM * 4));
  half_t* H1   = (half_t*)(ws + carve((size_t)MCHUNK * HDIM * 2));
  (void)ws_size; (void)in_sizes; (void)n_in; (void)out_size;

  // 1. convert + transpose weights to f16
  convT_kernel<<<(192 * 576 + 255) / 256, 256, 0, stream>>>(qkv_w, qkvT, 192, 576);
  convT_kernel<<<(192 * 192 + 255) / 256, 256, 0, stream>>>(proj_w, projT, 192, 192);
  convT_kernel<<<(192 * 768 + 255) / 256, 256, 0, stream>>>(fc1_w, fc1T, 192, 768);
  convT_kernel<<<(768 * 192 + 255) / 256, 256, 0, stream>>>(fc2_w, fc2T, 768, 192);

  // 2. LN1 + roll + window partition -> A1 f16 (windowed token order)
  ln_kernel<<<LTOK / 8, 256, 0, stream>>>(x, n1g, n1b, A1, 1);

  // 3. QKV GEMM [L,192]@[192,576] -> q(scaled)/k/v per (window,head)
  gemm128_kernel<0><<<dim3(LTOK / 128, 576 / 64), 128, 0, stream>>>(
      A1, qkvT, qkv_b, 192, nullptr, qbuf, kbuf, vbuf, nullptr, 0);

  // 4. windowed attention (A1 reused as attention output [L,192] f16)
  attn_kernel<<<NLON * NW * NHEAD, 128, 0, stream>>>(qbuf, kbuf, vbuf, btab, A1);

  // 5. proj GEMM + un-window + roll + residual -> xmid f32
  gemm128_kernel<1><<<dim3(LTOK / 128, 192 / 64), 128, 0, stream>>>(
      A1, projT, proj_b, 192, xmid, nullptr, nullptr, nullptr, x, 0);

  // 6. LN2 -> A2 f16 (reuse qbuf)
  half_t* A2 = qbuf;
  ln_kernel<<<LTOK / 8, 256, 0, stream>>>(xmid, n2g, n2b, A2, 0);

  // 7. MLP, chunked so the 768-wide hidden reuses one small buffer.
  for (int c = 0; c < 5; ++c) {
    int m0 = c * MCHUNK;
    gemm128_kernel<2><<<dim3(MCHUNK / 128, 768 / 64), 128, 0, stream>>>(
        A2 + (size_t)m0 * CDIM, fc1T, fc1_b, 192, nullptr, H1, nullptr, nullptr,
        nullptr, 0);
    gemm128_kernel<3><<<dim3(MCHUNK / 128, 192 / 64), 128, 0, stream>>>(
        H1, fc2T, fc2_b, 768, out, nullptr, nullptr, nullptr, xmid, m0);
  }
}
